// Courbariaux_Binary_Net_MNIST_7971459301381
// MI455X (gfx1250) — 8-bit matrix operand order repaired and checked by reading, compiled, not run on hardware
//
#include <hip/hip_runtime.h>

typedef int v8i __attribute__((ext_vector_type(8)));

#define BATCH 16384
#define DIM   1024
#define NCLS  10

static __device__ __forceinline__ v8i wmma_iu8(v8i a, v8i b, v8i c) {
  return __builtin_amdgcn_wmma_i32_16x16x64_iu8(true, a, true, b, c, false, false);
}


__global__ __launch_bounds__(256) void k_bin_input(const float* __restrict__ x,
                                                   signed char* __restrict__ o, int n4) {
  int i = blockIdx.x * blockDim.x + threadIdx.x;
  if (i >= n4) return;
  float4 v = ((const float4*)x)[i];
  char4 c;
  c.x = (v.x >= 0.5f) ? 1 : -1;
  c.y = (v.y >= 0.5f) ? 1 : -1;
  c.z = (v.z >= 0.5f) ? 1 : -1;
  c.w = (v.w >= 0.5f) ? 1 : -1;
  ((char4*)o)[i] = c;
}

__global__ __launch_bounds__(256) void k_bin_w(const float* __restrict__ w,
                                               signed char* __restrict__ o, int n4) {
  int i = blockIdx.x * blockDim.x + threadIdx.x;
  if (i >= n4) return;
  float4 v = ((const float4*)w)[i];
  char4 c;
  c.x = (v.x >= 0.f) ? 1 : -1;
  c.y = (v.y >= 0.f) ? 1 : -1;
  c.z = (v.z >= 0.f) ? 1 : -1;
  c.w = (v.w >= 0.f) ? 1 : -1;
  ((char4*)o)[i] = c;
}

__global__ __launch_bounds__(256) void k_bin_w4(const float* __restrict__ w,
                                                signed char* __restrict__ o) {
  int i = blockIdx.x * blockDim.x + threadIdx.x;
  int row = i >> 10;
  signed char v = 0;
  if (row < NCLS) v = (w[i] >= 0.f) ? 1 : -1;
  o[i] = v;
}


__global__ __launch_bounds__(256) void k_bnn_hidden(const signed char* __restrict__ Ain,
                                                    const signed char* __restrict__ Wb,
                                                    const float* __restrict__ gg,
                                                    const float* __restrict__ bb,
                                                    const float* __restrict__ mm,
                                                    const float* __restrict__ vv,
                                                    signed char* __restrict__ Aout) {
  const int lane = threadIdx.x & 31;
  const int lh   = lane & 15;
  const int hi   = lane >> 4;
  const int wid  = blockIdx.x * (blockDim.x >> 5) + (threadIdx.x >> 5);
  const int NT   = DIM / 32;
  const int m0   = (wid / NT) * 64;
  const int n0   = (wid % NT) * 32;

  const signed char* aBase0 = Ain + (size_t)(m0 +  0 + lh) * DIM + hi * 8;
  const signed char* aBase1 = Ain + (size_t)(m0 + 16 + lh) * DIM + hi * 8;
  const signed char* aBase2 = Ain + (size_t)(m0 + 32 + lh) * DIM + hi * 8;
  const signed char* aBase3 = Ain + (size_t)(m0 + 48 + lh) * DIM + hi * 8;
  const signed char* bBase0 = Wb + (size_t)(n0 +  0 + lh) * DIM + hi * 8;
  const signed char* bBase1 = Wb + (size_t)(n0 + 16 + lh) * DIM + hi * 8;

  v8i acc[4][2];
#pragma unroll
  for (int t = 0; t < 4; ++t)
#pragma unroll
    for (int j = 0; j < 2; ++j)
      acc[t][j] = (v8i){0, 0, 0, 0, 0, 0, 0, 0};

  for (int kb = 0; kb < DIM; kb += 64) {
    v8i A[4], B[2];
    const signed char* ab[4] = {aBase0, aBase1, aBase2, aBase3};
#pragma unroll
    for (int t = 0; t < 4; ++t) {
      int2 a0 = *(const int2*)(ab[t] + kb + 0);
      int2 a1 = *(const int2*)(ab[t] + kb + 16);
      int2 a2 = *(const int2*)(ab[t] + kb + 32);
      int2 a3 = *(const int2*)(ab[t] + kb + 48);
      A[t] = (v8i){a0.x, a0.y, a1.x, a1.y, a2.x, a2.y, a3.x, a3.y};
    }
    const signed char* bbp[2] = {bBase0, bBase1};
#pragma unroll
    for (int j = 0; j < 2; ++j) {
      int2 b0 = *(const int2*)(bbp[j] + kb + 0);
      int2 b1 = *(const int2*)(bbp[j] + kb + 16);
      int2 b2 = *(const int2*)(bbp[j] + kb + 32);
      int2 b3 = *(const int2*)(bbp[j] + kb + 48);
      B[j] = (v8i){b0.x, b0.y, b1.x, b1.y, b2.x, b2.y, b3.x, b3.y};
    }
#pragma unroll
    for (int t = 0; t < 4; ++t)
#pragma unroll
      for (int j = 0; j < 2; ++j)
        acc[t][j] = wmma_iu8(A[t], B[j], acc[t][j]);
  }

#pragma unroll
  for (int j = 0; j < 2; ++j) {
    const int n = n0 + 16 * j + lh;
    const float sc = gg[n] * rsqrtf(vv[n] + 1e-5f);
    const float sh = bb[n] - mm[n] * sc;
#pragma unroll
    for (int t = 0; t < 4; ++t) {
      const int mrow = m0 + 16 * t + 8 * hi;
#pragma unroll
      for (int r = 0; r < 8; ++r) {
        float y = (float)acc[t][j][r] * sc + sh;
        Aout[(size_t)(mrow + r) * DIM + n] = (y >= 0.f) ? 1 : -1;
      }
    }
  }
}


__global__ __launch_bounds__(256) void k_bnn_final(const signed char* __restrict__ Ain,
                                                   const signed char* __restrict__ Wb4,
                                                   const float* __restrict__ tn_w,
                                                   const float* __restrict__ tn_b,
                                                   const float* __restrict__ tn_m,
                                                   const float* __restrict__ tn_v,
                                                   float* __restrict__ out) {
  const int lane = threadIdx.x & 31;
  const int lh   = lane & 15;
  const int hi   = lane >> 4;
  const int wid  = blockIdx.x * (blockDim.x >> 5) + (threadIdx.x >> 5);
  const int m0   = wid * 64;

  const signed char* aBase0 = Ain + (size_t)(m0 +  0 + lh) * DIM + hi * 8;
  const signed char* aBase1 = Ain + (size_t)(m0 + 16 + lh) * DIM + hi * 8;
  const signed char* aBase2 = Ain + (size_t)(m0 + 32 + lh) * DIM + hi * 8;
  const signed char* aBase3 = Ain + (size_t)(m0 + 48 + lh) * DIM + hi * 8;
  const signed char* bBase  = Wb4 + (size_t)lh * DIM + hi * 8;

  v8i acc[4];
#pragma unroll
  for (int t = 0; t < 4; ++t) acc[t] = (v8i){0, 0, 0, 0, 0, 0, 0, 0};

  for (int kb = 0; kb < DIM; kb += 64) {
    v8i A[4], B;
    const signed char* ab[4] = {aBase0, aBase1, aBase2, aBase3};
#pragma unroll
    for (int t = 0; t < 4; ++t) {
      int2 a0 = *(const int2*)(ab[t] + kb + 0);
      int2 a1 = *(const int2*)(ab[t] + kb + 16);
      int2 a2 = *(const int2*)(ab[t] + kb + 32);
      int2 a3 = *(const int2*)(ab[t] + kb + 48);
      A[t] = (v8i){a0.x, a0.y, a1.x, a1.y, a2.x, a2.y, a3.x, a3.y};
    }
    int2 b0 = *(const int2*)(bBase + kb + 0);
    int2 b1 = *(const int2*)(bBase + kb + 16);
    int2 b2 = *(const int2*)(bBase + kb + 32);
    int2 b3 = *(const int2*)(bBase + kb + 48);
    B = (v8i){b0.x, b0.y, b1.x, b1.y, b2.x, b2.y, b3.x, b3.y};
#pragma unroll
    for (int t = 0; t < 4; ++t) acc[t] = wmma_iu8(A[t], B, acc[t]);
  }

  const float s  = rsqrtf(tn_v[0] + 1e-4f) * tn_w[0];
  const float tm = tn_m[0];
  const float tb = tn_b[0];
  const int n = lh;
  if (n < NCLS) {
#pragma unroll
    for (int t = 0; t < 4; ++t) {
      const int mrow = m0 + 16 * t + 8 * hi;
#pragma unroll
      for (int r = 0; r < 8; ++r)
        out[(size_t)(mrow + r) * NCLS + n] = ((float)acc[t][r] - tm) * s + tb;
    }
  }
}


extern "C" void kernel_launch(void* const* d_in, const int* in_sizes, int n_in,
                              void* d_out, int out_size, void* d_ws, size_t ws_size,
                              hipStream_t stream) {
  const float* x  = (const float*)d_in[0];
  const float* W1 = (const float*)d_in[1];
  const float* W2 = (const float*)d_in[2];
  const float* W3 = (const float*)d_in[3];
  const float* W4 = (const float*)d_in[4];
  const float* g1 = (const float*)d_in[5],  *b1 = (const float*)d_in[6];
  const float* m1 = (const float*)d_in[7],  *v1 = (const float*)d_in[8];
  const float* g2 = (const float*)d_in[9],  *b2 = (const float*)d_in[10];
  const float* m2 = (const float*)d_in[11], *v2 = (const float*)d_in[12];
  const float* g3 = (const float*)d_in[13], *b3 = (const float*)d_in[14];
  const float* m3 = (const float*)d_in[15], *v3 = (const float*)d_in[16];
  const float* tn_w = (const float*)d_in[17];
  const float* tn_b = (const float*)d_in[18];
  const float* tn_m = (const float*)d_in[19];
  const float* tn_v = (const float*)d_in[20];
  float* out = (float*)d_out;

  signed char* A0  = (signed char*)d_ws;
  signed char* A1  = A0  + (size_t)BATCH * DIM;
  signed char* W1b = A1  + (size_t)BATCH * DIM;
  signed char* W2b = W1b + (size_t)DIM * DIM;
  signed char* W3b = W2b + (size_t)DIM * DIM;
  signed char* W4b = W3b + (size_t)DIM * DIM;

  {
    int n4 = BATCH * DIM / 4;
    k_bin_input<<<n4 / 256, 256, 0, stream>>>(x, A0, n4);
  }
  {
    int n4 = DIM * DIM / 4;
    k_bin_w<<<n4 / 256, 256, 0, stream>>>(W1, W1b, n4);
    k_bin_w<<<n4 / 256, 256, 0, stream>>>(W2, W2b, n4);
    k_bin_w<<<n4 / 256, 256, 0, stream>>>(W3, W3b, n4);
  }
  k_bin_w4<<<16 * DIM / 256, 256, 0, stream>>>(W4, W4b);

  const int hidden_blocks = (BATCH / 64) * (DIM / 32) / 8;
  k_bnn_hidden<<<hidden_blocks, 256, 0, stream>>>(A0, W1b, g1, b1, m1, v1, A1);
  k_bnn_hidden<<<hidden_blocks, 256, 0, stream>>>(A1, W2b, g2, b2, m2, v2, A0);
  k_bnn_hidden<<<hidden_blocks, 256, 0, stream>>>(A0, W3b, g3, b3, m3, v3, A1);

  k_bnn_final<<<(BATCH / 64) / 8, 256, 0, stream>>>(A1, W4b, tn_w, tn_b, tn_m, tn_v, out);
}
